// MultiHeadedDotAttention_63050119905518
// MI455X (gfx1250) — matrix operand order repaired and checked by reading, not compiled, not run on hardware
//
#include <hip/hip_runtime.h>
#include <math.h>
#include <stdint.h>

typedef __attribute__((ext_vector_type(16))) _Float16 v16h;
typedef __attribute__((ext_vector_type(8)))  _Float16 v8h;
typedef __attribute__((ext_vector_type(8)))  float    v8f;

typedef __attribute__((ext_vector_type(4))) unsigned tdmv4u;
typedef __attribute__((ext_vector_type(8))) int      tdmv8i;
typedef __attribute__((ext_vector_type(4))) int      tdmv4i;

constexpr int B  = 4;
constexpr int S  = 1024;
constexpr int D  = 1024;
constexpr int H  = 16;
constexpr int DK = 64;
constexpr long NX = (long)B * S * D;   // 4,194,304 elements
constexpr long NW = (long)D * D;       // 1,048,576 elements

// Build a 16x32 f16 A-fragment (contiguous K order, matching B):
// lanes 0-15 (hs=0): K = k0+0..15
// lanes 16-31 (hs=1): K = k0+16..31
__device__ __forceinline__ v16h make_a_frag(const _Float16* rowptr, int k0, int hs) {
  union { v16h v; v8h h[2]; } u;
  u.h[0] = *(const v8h*)(rowptr + k0 + hs * 16);
  u.h[1] = *(const v8h*)(rowptr + k0 + hs * 16 + 8);
  return u.v;
}

// ---- Tensor Data Mover helpers (ISA cdna5 §8: D# groups in SGPRs) ----
// group0: [31:0]=count=1(user), [63:32]=lds_addr, [120:64]=global_addr, [127:126]=type=2
__device__ __forceinline__ void tdm_load_to_lds(unsigned lds_addr, unsigned long long ga,
                                                tdmv8i g1) {
  tdmv4u g0;
  g0[0] = 1u;                       // count=1, is_restore=0, gather off
  g0[1] = lds_addr;                 // LDS byte address
  g0[2] = (unsigned)ga;             // global_addr[31:0]
  g0[3] = ((unsigned)(ga >> 32) & 0x01FFFFFFu) | 0x80000000u;  // addr[56:32] | type=2
  tdmv4i z = (tdmv4i)0;             // groups 2/3 unused (<=3D handled in g1)
#if defined(__clang_major__) && (__clang_major__ >= 23)
  __builtin_amdgcn_tensor_load_to_lds(g0, g1, z, z, (tdmv8i)0, 0);
#else
  __builtin_amdgcn_tensor_load_to_lds(g0, g1, z, z, 0);
#endif
}

// group1 for a contiguous 1-D run of `nelem` f16 elements
__device__ __forceinline__ tdmv8i tdm_g1_1d(unsigned nelem) {
  tdmv8i g = (tdmv8i)0;
  const unsigned td0 = 1u << 20;                 // tensor_dim0 (large, in-bounds)
  g[0] = 0x00010000;                             // workgroup_mask=0, data_size=1 (2B)
  g[1] = (int)((td0 & 0xFFFFu) << 16);           // bits 63:48 = tensor_dim0[15:0]
  g[2] = (int)(((td0 >> 16) & 0xFFFFu)           // bits 79:64 = tensor_dim0[31:16]
               | (1u << 16));                    // bits 95:80 = tensor_dim1[15:0] = 1
  g[3] = (int)(nelem << 16);                     // bits 127:112 = tile_dim0
  g[4] = 0;                                      // tile_dim1=0 (unused), tile_dim2=0
  g[5] = (int)td0;                               // tensor_dim0_stride[31:0]
  return g;
}

// group1 for a 2-D tile: tile_dim0 x tile_dim1 f16 elements, row stride `stride0`
__device__ __forceinline__ tdmv8i tdm_g1_2d(unsigned tile0, unsigned tile1, unsigned stride0) {
  tdmv8i g = (tdmv8i)0;
  const unsigned td0 = 1u << 20, td1 = 1u << 20;
  g[0] = 0x00010000;                             // data_size = 2 bytes
  g[1] = (int)((td0 & 0xFFFFu) << 16);
  g[2] = (int)(((td0 >> 16) & 0xFFFFu) | ((td1 & 0xFFFFu) << 16));
  g[3] = (int)(((td1 >> 16) & 0xFFFFu) | (tile0 << 16));
  g[4] = (int)(tile1 & 0xFFFFu);                 // tile_dim1; tile_dim2=0
  g[5] = (int)stride0;                           // tensor_dim0_stride[31:0]
  return g;
}

__global__ void cvt_f32_f16(const float* __restrict__ src, _Float16* __restrict__ dst, int n) {
  int i = blockIdx.x * blockDim.x + threadIdx.x;
  if (i < n) dst[i] = (_Float16)src[i];
}

// C = X @ W^T + bias, X:[M_TOK,D] f16 row-major, W:[D,D] f16 row-major (N x K).
// One wave computes a 64x64 tile with 4x4 v8f accumulators.
// transpose_out==0 -> out[b][h][s][dk]   (Q, K)
// transpose_out==1 -> out[b][h][dk][s]   (V transposed for P@V B-fragments)
__global__ __launch_bounds__(256)
void qkv_gemm_kernel(const _Float16* __restrict__ X,
                     const _Float16* __restrict__ W,
                     const float*    __restrict__ bias,
                     _Float16*       __restrict__ out,
                     int transpose_out) {
  const int wave = blockIdx.x * 8 + (threadIdx.x >> 5);
  const int tilesN = D / 64;            // 16
  const int tm = wave / tilesN;         // 0..63
  const int tn = wave % tilesN;         // 0..15
  const int lane = threadIdx.x & 31;
  const int r  = lane & 15;
  const int hs = lane >> 4;

  const int Mbase = tm * 64;
  const int Nbase = tn * 64;

  const _Float16* arow[4];
  const _Float16* brow[4];
#pragma unroll
  for (int i = 0; i < 4; ++i) {
    arow[i] = X + (size_t)(Mbase + i * 16 + r) * D;
    brow[i] = W + (size_t)(Nbase + i * 16 + r) * D;   // B column n = W row n
  }

  v8f acc[4][4];
#pragma unroll
  for (int mi = 0; mi < 4; ++mi)
#pragma unroll
    for (int ni = 0; ni < 4; ++ni) acc[mi][ni] = (v8f)0.0f;

  for (int k0 = 0; k0 < D; k0 += 32) {
    v16h a[4], b[4];
#pragma unroll
    for (int mi = 0; mi < 4; ++mi) a[mi] = make_a_frag(arow[mi], k0, hs);
#pragma unroll
    for (int ni = 0; ni < 4; ++ni)
      b[ni] = *(const v16h*)(brow[ni] + k0 + hs * 16);  // lanes 0-15: K 0..15; 16-31: K 16..31
#pragma unroll
    for (int mi = 0; mi < 4; ++mi)
#pragma unroll
      for (int ni = 0; ni < 4; ++ni)
        acc[mi][ni] = __builtin_amdgcn_wmma_f32_16x16x32_f16(
            false, a[mi], false, b[ni], (short)0, acc[mi][ni], false, false);
  }

  // Epilogue: bias + f16 store with head split / optional transpose.
#pragma unroll
  for (int mi = 0; mi < 4; ++mi) {
#pragma unroll
    for (int ni = 0; ni < 4; ++ni) {
      const int n  = Nbase + ni * 16 + r;   // feature (C layout: N = lane&15)
      const float bn = bias[n];
      const int h  = n >> 6;                // n / 64
      const int dk = n & 63;
#pragma unroll
      for (int v = 0; v < 8; ++v) {
        const int m  = Mbase + mi * 16 + hs * 8 + v;  // token (C layout: M = hs*8 + v)
        const int bb = m >> 10;                       // m / S
        const int s  = m & (S - 1);
        const _Float16 val = (_Float16)(acc[mi][ni][v] + bn);
        const size_t off = transpose_out
            ? (((size_t)(bb * H + h) * DK + dk) * S + s)
            : (((size_t)(bb * H + h) * S + s) * DK + dk);
        out[off] = val;
      }
    }
  }
}

// Flash attention: one wave handles (b, h, 16 query rows). K/V tiles are staged
// into double-buffered LDS by the Tensor Data Mover (TENSORcnt-tracked DMA):
// tile i+1 is prefetched, then s_wait_tensorcnt(2) guarantees tile i landed
// (TDM ops from one wave complete in order), overlapping DMA with softmax+WMMA.
// Buffer selection is pure integer-offset arithmetic on one LDS pointer chain
// so clang keeps the loads in addrspace(3) (ds_load, not flat).
__global__ __launch_bounds__(256)
void attn_kernel(const _Float16* __restrict__ Q,   // [B,H,S,DK]
                 const _Float16* __restrict__ K,   // [B,H,S,DK]
                 const _Float16* __restrict__ Vt,  // [B,H,DK,S]
                 const int* __restrict__ mask,     // [B,S]
                 float* __restrict__ out) {        // [B,S,D]
  // per wave: 512 P-staging + 2x(2048 K + 2048 V) halves = 17KB; 8 waves = 136KB
  __shared__ _Float16 smem[8 * 8704];
  const int wib  = threadIdx.x >> 5;
  const int wave = blockIdx.x * 8 + wib;
  const int qt = wave & (S / 16 - 1);              // 64 query tiles
  const int bh = wave / (S / 16);
  const int h  = bh & (H - 1);
  const int b  = bh / H;
  const int lane = threadIdx.x & 31;
  const int r  = lane & 15;
  const int hs = lane >> 4;
  const int qbase = qt * 16;

  // Per-wave layout (halves): [0,512) P | [512,2560) K0 | [2560,4608) V0
  //                           [4608,6656) K1 | [6656,8704) V1
  _Float16* plds = smem + wib * 8704;
  // LDS byte offsets for the TDM D#: generic LDS pointers carry the aperture in
  // the high bits; low 32 bits are the in-LDS byte address.
  const unsigned base_off = (unsigned)(uintptr_t)plds;

  const _Float16* Qbh = Q  + (size_t)(b * H + h) * S * DK;
  const _Float16* Kbh = K  + (size_t)(b * H + h) * S * DK;
  const _Float16* Vbh = Vt + (size_t)(b * H + h) * DK * S;
  const int* mb = mask + b * S;

  const tdmv8i g1k = tdm_g1_1d(32 * DK);           // K tile: contiguous 2048 halves
  const tdmv8i g1v = tdm_g1_2d(32, DK, S);         // V tile: 32 x 64, row stride S

  // Fixed Q fragments (16x64 split into two 16x32 A-fragments)
  const _Float16* qrow = Qbh + (size_t)(qbase + r) * DK;
  const v16h aq0 = make_a_frag(qrow, 0, hs);
  const v16h aq1 = make_a_frag(qrow, 32, hs);

  v8f o[4];
#pragma unroll
  for (int t = 0; t < 4; ++t) o[t] = (v8f)0.0f;
  float mrow[8], lrow[8];
#pragma unroll
  for (int v = 0; v < 8; ++v) { mrow[v] = -1e30f; lrow[v] = 0.0f; }

  const float scale = 0.125f;       // 1/sqrt(DK)
  const int kend = qbase + 16;      // causal: keys < kend

  // ---- TDM pipeline prologue: DMA tile 0 into buffer 0 ----
  tdm_load_to_lds(base_off + 512 * 2, (unsigned long long)(uintptr_t)Kbh, g1k);
  tdm_load_to_lds(base_off + 2560 * 2, (unsigned long long)(uintptr_t)Vbh, g1v);
  int cur = 0;

  for (int kb = 0; kb < kend; kb += 32) {
    const int nxt = cur ^ 1;
    if (kb + 32 < kend) {
      // Prefetch tile i+1, then let exactly those 2 TDM ops stay outstanding.
      tdm_load_to_lds(base_off + (512 + nxt * 4096) * 2,
                      (unsigned long long)(uintptr_t)(Kbh + (size_t)(kb + 32) * DK), g1k);
      tdm_load_to_lds(base_off + (2560 + nxt * 4096) * 2,
                      (unsigned long long)(uintptr_t)(Vbh + kb + 32), g1v);
      __builtin_amdgcn_s_wait_tensorcnt(2);
    } else {
      __builtin_amdgcn_s_wait_tensorcnt(0);
    }
    const int co = cur * 4096;                     // half-element offset of live buffer
    const _Float16* kt = plds + 512 + co;
    const _Float16* vt = plds + 2560 + co;

    // ---- scores: two 16x16 fragments (keys kb..+15, kb+16..+31) ----
    float sc[2][8];
#pragma unroll
    for (int nt = 0; nt < 2; ++nt) {
      const int key = kb + nt * 16 + r;            // B column = key row
      const _Float16* krow = kt + (nt * 16 + r) * DK;
      const v16h bk0 = *(const v16h*)(krow + hs * 16);
      const v16h bk1 = *(const v16h*)(krow + 32 + hs * 16);
      v8f sacc = (v8f)0.0f;
      sacc = __builtin_amdgcn_wmma_f32_16x16x32_f16(false, aq0, false, bk0, (short)0, sacc, false, false);
      sacc = __builtin_amdgcn_wmma_f32_16x16x32_f16(false, aq1, false, bk1, (short)0, sacc, false, false);
      const int am = mb[key];
#pragma unroll
      for (int v = 0; v < 8; ++v) {
        const int q = qbase + hs * 8 + v;
        float x = sacc[v] * scale;
        if (key > q || am == 0) x = -1e30f;        // causal + attention mask
        sc[nt][v] = x;
      }
    }
    // ---- online softmax (row reductions across 16-lane halves) ----
    float ef[8];
#pragma unroll
    for (int v = 0; v < 8; ++v) {
      float mx = fmaxf(sc[0][v], sc[1][v]);
#pragma unroll
      for (int off = 8; off >= 1; off >>= 1)
        mx = fmaxf(mx, __shfl_xor(mx, off, 32));
      const float mnew = fmaxf(mrow[v], mx);
      ef[v] = __expf(mrow[v] - mnew);
      const float p0 = __expf(sc[0][v] - mnew);
      const float p1 = __expf(sc[1][v] - mnew);
      sc[0][v] = p0; sc[1][v] = p1;
      float sum = p0 + p1;
#pragma unroll
      for (int off = 8; off >= 1; off >>= 1)
        sum += __shfl_xor(sum, off, 32);
      lrow[v] = lrow[v] * ef[v] + sum;
      mrow[v] = mnew;
    }
#pragma unroll
    for (int t = 0; t < 4; ++t)
#pragma unroll
      for (int v = 0; v < 8; ++v) o[t][v] *= ef[v];

    // ---- C-layout -> A-layout transpose of P through wave-private LDS ----
#pragma unroll
    for (int v = 0; v < 8; ++v) {
      plds[(hs * 8 + v) * 32 + r]      = (_Float16)sc[0][v];
      plds[(hs * 8 + v) * 32 + 16 + r] = (_Float16)sc[1][v];
    }
    asm volatile("s_wait_dscnt 0" ::: "memory");   // CDNA5 split DS counter wait
    union { v16h v; v8h hh[2]; } up;
    up.hh[0] = *(const v8h*)(plds + r * 32 + hs * 16);
    up.hh[1] = *(const v8h*)(plds + r * 32 + hs * 16 + 8);
    const v16h ap = up.v;

    // ---- P @ V from the LDS V tile (row dk = 32 halves) ----
#pragma unroll
    for (int t = 0; t < 4; ++t) {
      const int dk = t * 16 + r;
      const v16h bv = *(const v16h*)(vt + dk * 32 + hs * 16);
      o[t] = __builtin_amdgcn_wmma_f32_16x16x32_f16(false, ap, false, bv, (short)0, o[t], false, false);
    }
    cur = nxt;
  }

  // ---- normalize + write fp32 output [B,S,H*DK] ----
#pragma unroll
  for (int t = 0; t < 4; ++t) {
    const int dk = t * 16 + r;
#pragma unroll
    for (int v = 0; v < 8; ++v) {
      const int q = qbase + hs * 8 + v;
      out[(size_t)(b * S + q) * D + h * DK + dk] = o[t][v] / lrow[v];
    }
  }
}

extern "C" void kernel_launch(void* const* d_in, const int* in_sizes, int n_in,
                              void* d_out, int out_size, void* d_ws, size_t ws_size,
                              hipStream_t stream) {
  const float* query = (const float*)d_in[0];
  const float* key_  = (const float*)d_in[1];
  const float* value = (const float*)d_in[2];
  const int*   amask = (const int*)d_in[3];
  const float* Wq = (const float*)d_in[4];
  const float* bq = (const float*)d_in[5];
  const float* Wk = (const float*)d_in[6];
  const float* bk = (const float*)d_in[7];
  const float* Wv = (const float*)d_in[8];
  const float* bv = (const float*)d_in[9];
  float* out = (float*)d_out;

  // Workspace layout (f16 halves): ~56.6 MB total
  _Float16* w   = (_Float16*)d_ws;
  _Float16* qh  = w;
  _Float16* kh  = qh + NX;
  _Float16* vh  = kh + NX;
  _Float16* wqh = vh + NX;
  _Float16* wkh = wqh + NW;
  _Float16* wvh = wkh + NW;
  _Float16* Qh  = wvh + NW;
  _Float16* Kh  = Qh + NX;
  _Float16* Vth = Kh + NX;

  // Stage 1: f32 -> f16 conversions
  cvt_f32_f16<<<(int)((NX + 255) / 256), 256, 0, stream>>>(query, qh, (int)NX);
  cvt_f32_f16<<<(int)((NX + 255) / 256), 256, 0, stream>>>(key_,  kh, (int)NX);
  cvt_f32_f16<<<(int)((NX + 255) / 256), 256, 0, stream>>>(value, vh, (int)NX);
  cvt_f32_f16<<<(int)((NW + 255) / 256), 256, 0, stream>>>(Wq, wqh, (int)NW);
  cvt_f32_f16<<<(int)((NW + 255) / 256), 256, 0, stream>>>(Wk, wkh, (int)NW);
  cvt_f32_f16<<<(int)((NW + 255) / 256), 256, 0, stream>>>(Wv, wvh, (int)NW);

  // Stage 2: QKV projections. 1024 wave-tiles of 64x64, 8 waves/block -> 128 blocks.
  qkv_gemm_kernel<<<128, 256, 0, stream>>>(qh, wqh, bq, Qh, 0);
  qkv_gemm_kernel<<<128, 256, 0, stream>>>(kh, wkh, bk, Kh, 0);
  qkv_gemm_kernel<<<128, 256, 0, stream>>>(vh, wvh, bv, Vth, 1);

  // Stage 3: attention. B*H*(S/16) = 4096 wave-units, 8 waves/block -> 512 blocks.
  attn_kernel<<<512, 256, 0, stream>>>(Qh, Kh, Vth, amask, out);
}
